// MultiHeadAttention_63385127354958
// MI455X (gfx1250) — matrix operand order repaired and checked by reading, not compiled, not run on hardware
//
#include <hip/hip_runtime.h>
#include <math.h>

#define B_  2
#define S_  2048
#define D_  1024
#define H_  16
#define DK_ 64
#define DV_ 64
#define DO_ 1024

typedef __bf16 bf16;
typedef __attribute__((ext_vector_type(16))) __bf16 v16bf;
typedef __attribute__((ext_vector_type(8)))  __bf16 v8bf;
typedef __attribute__((ext_vector_type(8)))  float  v8f;

#define WMMA_BF16(A, B, C) \
  __builtin_amdgcn_wmma_f32_16x16x32_bf16(false, (A), false, (B), (short)0, (C), false, false)

static __device__ __forceinline__ v8f zero8() { v8f z = {}; return z; }

// A fragment (16x32 bf16, M x K): lane holds row r=lane&15; cols g*8..g*8+7 and
// g*8+16..g*8+23 where g = lane>>4.  base points at [mbase, kbase]; ld in elements.
static __device__ __forceinline__ v16bf wmma_ld_a(const bf16* base, int ld, int lane) {
  const int r = lane & 15, g = lane >> 4;
  const bf16* p = base + (size_t)r * ld + g * 8;
  v8bf lo = *(const v8bf*)(p);
  v8bf hi = *(const v8bf*)(p + 16);
  v16bf a;
#pragma unroll
  for (int i = 0; i < 8; ++i) { a[i] = lo[i]; a[i + 8] = hi[i]; }
  return a;
}

// B fragment (32x16 bf16, K x N): lane holds column n=lane&15; k = g*8..g*8+7 and g*8+16..g*8+23.
// base points at element [nbase, kbase] of an N-major (transposed) matrix, ld in elems.
static __device__ __forceinline__ v16bf wmma_ld_b(const bf16* base, int ld, int lane) {
  const int n = lane & 15, g = lane >> 4;
  const bf16* p = base + (size_t)n * ld + g * 8;
  v8bf lo = *(const v8bf*)(p);
  v8bf hi = *(const v8bf*)(p + 16);
  v16bf b;
#pragma unroll
  for (int i = 0; i < 8; ++i) { b[i] = lo[i]; b[i + 8] = hi[i]; }
  return b;
}

// ---------------- stage 0: precision conversion / weight transposes ----------------
__global__ void k_cvt_bf16(const float* __restrict__ src, bf16* __restrict__ dst, int n) {
  for (int i = blockIdx.x * blockDim.x + threadIdx.x; i < n; i += gridDim.x * blockDim.x)
    dst[i] = (bf16)src[i];
}

// src: [Hn][Kn][Nn] f32 row-major  ->  dst: [Hn][Nn][Kn] bf16
__global__ void k_transpose_bf16(const float* __restrict__ src, bf16* __restrict__ dst,
                                 int Hn, int Kn, int Nn) {
  const int total = Hn * Kn * Nn;
  for (int i = blockIdx.x * blockDim.x + threadIdx.x; i < total; i += gridDim.x * blockDim.x) {
    const int h = i / (Kn * Nn);
    const int r = i - h * (Kn * Nn);
    const int k = r / Nn;
    const int n = r - k * Nn;
    dst[(size_t)h * Kn * Nn + (size_t)n * Kn + k] = (bf16)src[i];
  }
}

// ---------------- stage 1: fused QKV projection (V written transposed) ----------------
// per wave: 16 rows of x (bh-plane) x 64 cols for Q, K, V.  4096 waves total.
__global__ __launch_bounds__(256) void proj_kernel(
    const bf16* __restrict__ xb, const bf16* __restrict__ WqT,
    const bf16* __restrict__ WkT, const bf16* __restrict__ WvT,
    bf16* __restrict__ Qm, bf16* __restrict__ Km, bf16* __restrict__ VTm) {
  const int lane  = threadIdx.x & 31;
  const int gwave = blockIdx.x * 8 + (threadIdx.x >> 5);
  const int bh = gwave >> 7;   // 0..31
  const int mt = gwave & 127;  // q-row tile
  const int b = bh >> 4, h = bh & 15;

  const bf16* xr = xb + ((size_t)b * S_ + mt * 16) * D_;
  const bf16* wq = WqT + (size_t)h * DK_ * D_;
  const bf16* wk = WkT + (size_t)h * DK_ * D_;
  const bf16* wv = WvT + (size_t)h * DV_ * D_;

  v8f cq[4], ck[4], cv[4];
#pragma unroll
  for (int t = 0; t < 4; ++t) { cq[t] = zero8(); ck[t] = zero8(); cv[t] = zero8(); }

  for (int kk = 0; kk < D_; kk += 32) {
    __builtin_prefetch(xr + kk + 128, 0, 0);
    const v16bf a = wmma_ld_a(xr + kk, D_, lane);
#pragma unroll
    for (int t = 0; t < 4; ++t) {
      cq[t] = WMMA_BF16(a, wmma_ld_b(wq + (size_t)(t * 16) * D_ + kk, D_, lane), cq[t]);
      ck[t] = WMMA_BF16(a, wmma_ld_b(wk + (size_t)(t * 16) * D_ + kk, D_, lane), ck[t]);
      cv[t] = WMMA_BF16(a, wmma_ld_b(wv + (size_t)(t * 16) * D_ + kk, D_, lane), cv[t]);
    }
  }

  const int n = lane & 15, row0 = 8 * (lane >> 4);
#pragma unroll
  for (int v = 0; v < 8; ++v) {
    const int s = mt * 16 + row0 + v;
    const size_t qoff = ((size_t)bh * S_ + s) * DK_ + n;
#pragma unroll
    for (int t = 0; t < 4; ++t) {
      Qm[qoff + t * 16] = (bf16)cq[t][v];
      Km[qoff + t * 16] = (bf16)ck[t][v];
      VTm[((size_t)bh * DV_ + t * 16 + n) * S_ + s] = (bf16)cv[t][v];  // V transposed [64,S]
    }
  }
}

// ---------------- stage 2: flash attention, one wave = 16 query rows ----------------
__global__ __launch_bounds__(256) void attn_kernel(
    const bf16* __restrict__ Qm, const bf16* __restrict__ Km,
    const bf16* __restrict__ VTm, bf16* __restrict__ Om) {
  __shared__ __align__(16) bf16 ptile[8][16 * 32];  // per-wave P staging (16x32)
  const int lane  = threadIdx.x & 31;
  const int wv    = threadIdx.x >> 5;
  const int gwave = blockIdx.x * 8 + wv;
  const int bh = gwave >> 7;
  const int mt = gwave & 127;
  const int b = bh >> 4, h = bh & 15;

  const bf16* q  = Qm + ((size_t)bh * S_ + mt * 16) * DK_;
  const bf16* kp = Km + (size_t)bh * S_ * DK_;
  const bf16* vt = VTm + (size_t)bh * DV_ * S_;

  const v16bf aq0 = wmma_ld_a(q, DK_, lane);       // k-dims 0..31
  const v16bf aq1 = wmma_ld_a(q + 32, DK_, lane);  // k-dims 32..63

  float mrow[8], lrow[8];
  v8f oacc[4];
#pragma unroll
  for (int v = 0; v < 8; ++v) { mrow[v] = -1e30f; lrow[v] = 0.f; }
#pragma unroll
  for (int t = 0; t < 4; ++t) oacc[t] = zero8();

  const float sscale = 0.125f;  // 1/sqrt(64)
  const int col = lane & 15, row0 = 8 * (lane >> 4);

  for (int jj = 0; jj < S_; jj += 32) {
    // scores: two 16x16 tiles covering keys jj..jj+31
    const v16bf b00 = wmma_ld_b(kp + (size_t)jj * DK_,            DK_, lane);
    const v16bf b01 = wmma_ld_b(kp + (size_t)jj * DK_ + 32,       DK_, lane);
    const v16bf b10 = wmma_ld_b(kp + (size_t)(jj + 16) * DK_,     DK_, lane);
    const v16bf b11 = wmma_ld_b(kp + (size_t)(jj + 16) * DK_ + 32, DK_, lane);
    v8f s0 = zero8(), s1 = zero8();
    s0 = WMMA_BF16(aq0, b00, s0);
    s0 = WMMA_BF16(aq1, b01, s0);
    s1 = WMMA_BF16(aq0, b10, s1);
    s1 = WMMA_BF16(aq1, b11, s1);

    // online softmax per row (C layout: lane holds one key col, 8 rows)
#pragma unroll
    for (int v = 0; v < 8; ++v) {
      const float a0 = s0[v] * sscale;
      const float a1 = s1[v] * sscale;
      float t = fmaxf(a0, a1);
      t = fmaxf(t, __shfl_xor(t, 1, 32));
      t = fmaxf(t, __shfl_xor(t, 2, 32));
      t = fmaxf(t, __shfl_xor(t, 4, 32));
      t = fmaxf(t, __shfl_xor(t, 8, 32));
      const float mn = fmaxf(mrow[v], t);
      const float alpha = __expf(mrow[v] - mn);
      const float p0 = __expf(a0 - mn);
      const float p1 = __expf(a1 - mn);
      float ps = p0 + p1;
      ps += __shfl_xor(ps, 1, 32);
      ps += __shfl_xor(ps, 2, 32);
      ps += __shfl_xor(ps, 4, 32);
      ps += __shfl_xor(ps, 8, 32);
      lrow[v] = lrow[v] * alpha + ps;
      mrow[v] = mn;
      ptile[wv][(row0 + v) * 32 + col]      = (bf16)p0;
      ptile[wv][(row0 + v) * 32 + col + 16] = (bf16)p1;
#pragma unroll
      for (int t4 = 0; t4 < 4; ++t4) oacc[t4][v] *= alpha;
    }

    // re-shape P (C layout) -> A fragment via per-wave LDS tile
    asm volatile("s_wait_dscnt 0x0" ::: "memory");
    const v16bf ap = wmma_ld_a(&ptile[wv][0], 32, lane);
#pragma unroll
    for (int t4 = 0; t4 < 4; ++t4) {
      const v16bf bv = wmma_ld_b(vt + (size_t)(t4 * 16) * S_ + jj, S_, lane);
      oacc[t4] = WMMA_BF16(ap, bv, oacc[t4]);
    }
    asm volatile("s_wait_dscnt 0x0" ::: "memory");  // keep next iter's stores behind reads
  }

  // normalize + store concat-head layout [B,S,H*DV] in bf16
#pragma unroll
  for (int v = 0; v < 8; ++v) {
    const float inv = 1.0f / lrow[v];
    bf16* op = Om + ((size_t)b * S_ + mt * 16 + row0 + v) * (H_ * DV_) + h * DV_ + col;
#pragma unroll
    for (int t4 = 0; t4 < 4; ++t4) op[t4 * 16] = (bf16)(oacc[t4][v] * inv);
  }
}

// ---------------- stage 3: output projection [4096,1024] x [1024,1024] -> f32 ----------------
__global__ __launch_bounds__(256) void out_gemm(
    const bf16* __restrict__ Obf, const bf16* __restrict__ WoT, float* __restrict__ out) {
  const int lane  = threadIdx.x & 31;
  const int gwave = blockIdx.x * 8 + (threadIdx.x >> 5);
  const int mt  = gwave >> 4;  // 0..255 (16-row tiles over B*S)
  const int nt4 = gwave & 15;  // 64-col tiles over DO

  const bf16* ar = Obf + (size_t)mt * 16 * (H_ * DV_);
  const bf16* wo = WoT + (size_t)nt4 * 64 * (H_ * DV_);

  v8f c[4];
#pragma unroll
  for (int t = 0; t < 4; ++t) c[t] = zero8();

  for (int kk = 0; kk < H_ * DV_; kk += 32) {
    __builtin_prefetch(ar + kk + 128, 0, 0);
    const v16bf a = wmma_ld_a(ar + kk, H_ * DV_, lane);
#pragma unroll
    for (int t = 0; t < 4; ++t)
      c[t] = WMMA_BF16(a, wmma_ld_b(wo + (size_t)(t * 16) * (H_ * DV_) + kk, H_ * DV_, lane), c[t]);
  }

  const int n = lane & 15, row0 = 8 * (lane >> 4);
#pragma unroll
  for (int v = 0; v < 8; ++v) {
    float* op = out + (size_t)(mt * 16 + row0 + v) * DO_ + nt4 * 64 + n;
#pragma unroll
    for (int t = 0; t < 4; ++t) op[t * 16] = c[t][v];
  }
}

extern "C" void kernel_launch(void* const* d_in, const int* in_sizes, int n_in,
                              void* d_out, int out_size, void* d_ws, size_t ws_size,
                              hipStream_t stream) {
  const float* x  = (const float*)d_in[0];
  const float* Wq = (const float*)d_in[1];
  const float* Wk = (const float*)d_in[2];
  const float* Wv = (const float*)d_in[3];
  const float* Wo = (const float*)d_in[4];
  float* out = (float*)d_out;

  char* ws = (char*)d_ws;
  size_t off = 0;
  bf16* xb  = (bf16*)(ws + off); off += (size_t)B_ * S_ * D_ * sizeof(bf16);
  bf16* WqT = (bf16*)(ws + off); off += (size_t)H_ * DK_ * D_ * sizeof(bf16);
  bf16* WkT = (bf16*)(ws + off); off += (size_t)H_ * DK_ * D_ * sizeof(bf16);
  bf16* WvT = (bf16*)(ws + off); off += (size_t)H_ * DV_ * D_ * sizeof(bf16);
  bf16* WoT = (bf16*)(ws + off); off += (size_t)(H_ * DV_) * DO_ * sizeof(bf16);
  bf16* Qm  = (bf16*)(ws + off); off += (size_t)B_ * H_ * S_ * DK_ * sizeof(bf16);
  bf16* Km  = (bf16*)(ws + off); off += (size_t)B_ * H_ * S_ * DK_ * sizeof(bf16);
  bf16* VTm = (bf16*)(ws + off); off += (size_t)B_ * H_ * DV_ * S_ * sizeof(bf16);
  bf16* Obf = (bf16*)(ws + off); off += (size_t)B_ * S_ * H_ * DV_ * sizeof(bf16);
  (void)ws_size; (void)in_sizes; (void)n_in; (void)out_size;

  // stage 0: convert + transpose
  k_cvt_bf16<<<2048, 256, 0, stream>>>(x, xb, B_ * S_ * D_);
  k_transpose_bf16<<<1024, 256, 0, stream>>>(Wq, WqT, H_, D_, DK_);
  k_transpose_bf16<<<1024, 256, 0, stream>>>(Wk, WkT, H_, D_, DK_);
  k_transpose_bf16<<<1024, 256, 0, stream>>>(Wv, WvT, H_, D_, DV_);
  k_transpose_bf16<<<1024, 256, 0, stream>>>(Wo, WoT, 1, H_ * DV_, DO_);

  // stage 1: QKV projection — 4096 waves, 8 waves/WG
  proj_kernel<<<512, 256, 0, stream>>>(xb, WqT, WkT, WvT, Qm, Km, VTm);

  // stage 2: flash attention — 4096 waves
  attn_kernel<<<512, 256, 0, stream>>>(Qm, Km, VTm, Obf);

  // stage 3: output projection — 4096 waves
  out_gemm<<<512, 256, 0, stream>>>(Obf, WoT, out);
}
